// MMDLayer_60129542698
// MI455X (gfx1250) — matrix operand order repaired and checked by reading, not compiled, not run on hardware
//
#include <hip/hip_runtime.h>

#define N_ROWS 2048
#define DIM    64
#define NBATCH 32
#define ROW_STRIDE 36   // dwords per row in LDS (32 data + 4 pad; 144B = 9*16B aligned)
#define NCHUNK 16       // 2048 / 128
#define NPAIRS 136      // NCHUNK*(NCHUNK+1)/2

typedef __attribute__((ext_vector_type(16))) __bf16 v16bf;
typedef __attribute__((ext_vector_type(8)))  float  v8f;

#if __has_builtin(__builtin_amdgcn_exp2f)
#define EXP2F(x) __builtin_amdgcn_exp2f(x)   // raw v_exp_f32 (arg <= 0 here; tiny
#else                                        //  results flushing to 0 is fine)
#define EXP2F(x) exp2f(x)
#endif

union V16 {
  uint4 q[2];
  v16bf v;
};

// split two floats into packed-bf16 hi and lo dwords (low half = first element)
__device__ __forceinline__ void split2(float a, float b, uint& hi, uint& lo) {
  union { __bf16 h[2]; uint u; } ph, pl;
  const __bf16 ha = (__bf16)a, hb = (__bf16)b;
  ph.h[0] = ha; ph.h[1] = hb;
  pl.h[0] = (__bf16)(a - (float)ha);
  pl.h[1] = (__bf16)(b - (float)hb);
  hi = ph.u; lo = pl.u;
}

// cooperatively convert 128 rows (128x64 f32, contiguous) to bf16 hi/lo in LDS
__device__ __forceinline__ void stage_rows_bf16(const float* __restrict__ src,
                                                uint* ldsHi, uint* ldsLo, int tid) {
  for (int idx4 = tid; idx4 < (128 * DIM) / 4; idx4 += 256) {
    const float4 vv = ((const float4*)src)[idx4];
    const int g   = idx4 * 4;
    const int row = g >> 6;
    const int dw  = (g & 63) >> 1;        // dword index in row (2 bf16 per dword)
    uint h01, l01, h23, l23;
    split2(vv.x, vv.y, h01, l01);
    split2(vv.z, vv.w, h23, l23);
    const int base = row * ROW_STRIDE + dw;   // 8B aligned
    *(uint2*)&ldsHi[base] = make_uint2(h01, h23);
    *(uint2*)&ldsLo[base] = make_uint2(l01, l23);
  }
}

// ---------------------------------------------------------------------------
// Kernel 1: sq[b][i] = sum_d x^2  and  xy_acc[b] += sum_i exp(-sq/4)
// ---------------------------------------------------------------------------
__global__ __launch_bounds__(256) void mmd_sq_xy_kernel(
    const float* __restrict__ x, float* __restrict__ sq,
    float* __restrict__ xy_acc) {
  const int b   = blockIdx.y;
  const int row = blockIdx.x * 256 + threadIdx.x;

  const float* rp = x + ((size_t)b * N_ROWS + row) * DIM;
  float s = 0.f;
#pragma unroll
  for (int d = 0; d < DIM; d += 4) {
    float4 v = *(const float4*)(rp + d);
    s += v.x * v.x + v.y * v.y + v.z * v.z + v.w * v.w;
  }
  sq[b * N_ROWS + row] = s;

  float e = EXP2F(-0.36067376022224085f * s);   // exp(-s/4)
#pragma unroll
  for (int off = 16; off > 0; off >>= 1) e += __shfl_xor(e, off, 32);

  __shared__ float red[8];
  const int wave = threadIdx.x >> 5;
  const int lane = threadIdx.x & 31;
  if (lane == 0) red[wave] = e;
  __syncthreads();
  if (threadIdx.x == 0) {
    float t = 0.f;
#pragma unroll
    for (int w = 0; w < 8; ++w) t += red[w];
    atomicAdd(&xy_acc[b], t);
  }
}

// ---------------------------------------------------------------------------
// Kernel 2: symmetric gram via split-precision bf16 WMMA, fused exp2 -> yy_acc
// grid (136 chunk-pairs, 32 batches), block 256 (8 waves, 1 row-tile per wave)
// Block (I,J), I<=J: computes 128x128 element block once; weight 2 off-diag.
// ---------------------------------------------------------------------------
__global__ __launch_bounds__(256) void mmd_gram_exp_kernel(
    const float* __restrict__ x, const float* __restrict__ sq,
    float* __restrict__ yy_acc) {
  const int b = blockIdx.y;

  // triangular decode: pair index -> (I, J), I <= J
  int idx = blockIdx.x, I = 0;
  while (idx >= NCHUNK - I) { idx -= NCHUNK - I; ++I; }
  const int J = I + idx;

  alignas(16) __shared__ uint ldsHiA[128 * ROW_STRIDE];
  alignas(16) __shared__ uint ldsLoA[128 * ROW_STRIDE];
  alignas(16) __shared__ uint ldsHiB[128 * ROW_STRIDE];
  alignas(16) __shared__ uint ldsLoB[128 * ROW_STRIDE];
  __shared__ float sqRow[128];
  __shared__ float sqCol[128];
  __shared__ float red[8];

  const float* xb  = x  + (size_t)b * N_ROWS * DIM;
  const float* sqb = sq + b * N_ROWS;

  // --- stage both chunks as packed bf16 hi/lo ---
  stage_rows_bf16(xb + (size_t)I * 128 * DIM, ldsHiA, ldsLoA, threadIdx.x);
  stage_rows_bf16(xb + (size_t)J * 128 * DIM, ldsHiB, ldsLoB, threadIdx.x);
  if (threadIdx.x < 128) sqRow[threadIdx.x] = sqb[I * 128 + threadIdx.x];
  else                   sqCol[threadIdx.x - 128] = sqb[J * 128 + threadIdx.x - 128];
  __syncthreads();

  const int lane = threadIdx.x & 31;
  const int wave = threadIdx.x >> 5;
  const int r    = lane & 15;
  const int half = lane >> 4;

  // --- A operand registers: 2 x ds_load_b128 per chunk/precision ---
  // A dword index within row: 16c + 8*(v>>2) + 4*half + (v&3)
  v16bf ahi[2], alo[2];
  {
    const int abase = (wave * 16 + r) * ROW_STRIDE;
#pragma unroll
    for (int c = 0; c < 2; ++c) {
      V16 th, tl;
      th.q[0] = *(const uint4*)&ldsHiA[abase + 16 * c + 4 * half];
      th.q[1] = *(const uint4*)&ldsHiA[abase + 16 * c + 8 + 4 * half];
      tl.q[0] = *(const uint4*)&ldsLoA[abase + 16 * c + 4 * half];
      tl.q[1] = *(const uint4*)&ldsLoA[abase + 16 * c + 8 + 4 * half];
      ahi[c] = th.v;
      alo[c] = tl.v;
    }
  }

  // exp(-d2/2) = exp2(KE*(sqm+sqn) + L2E*g), clamped to arg <= 0
  const float L2E = 1.4426950408889634f;
  const float KE  = -0.5f * L2E;

  // C/D layout: c[v] holds M = v + 8*half, N = lane&15
  float sb[8];
#pragma unroll
  for (int v = 0; v < 8; ++v) sb[v] = KE * sqRow[wave * 16 + v + 8 * half];

  float acc = 0.f;

  // --- each wave: its row tile (from A chunk) x 8 column tiles (from B chunk)
#pragma unroll 2
  for (int ct = 0; ct < 8; ++ct) {
    const int bbase = (ct * 16 + r) * ROW_STRIDE;
    v8f cacc = {};
#pragma unroll
    for (int c = 0; c < 2; ++c) {
      // B dword index within row: 16c + 8*(v>>2) + 4*half + (v&3)
      V16 bh, bl;
      bh.q[0] = *(const uint4*)&ldsHiB[bbase + 16 * c + 4 * half];
      bh.q[1] = *(const uint4*)&ldsHiB[bbase + 16 * c + 8 + 4 * half];
      bl.q[0] = *(const uint4*)&ldsLoB[bbase + 16 * c + 4 * half];
      bl.q[1] = *(const uint4*)&ldsLoB[bbase + 16 * c + 8 + 4 * half];
      cacc = __builtin_amdgcn_wmma_f32_16x16x32_bf16(
          false, ahi[c], false, bh.v, (short)0, cacc, false, false);
      cacc = __builtin_amdgcn_wmma_f32_16x16x32_bf16(
          false, ahi[c], false, bl.v, (short)0, cacc, false, false);
      cacc = __builtin_amdgcn_wmma_f32_16x16x32_bf16(
          false, alo[c], false, bh.v, (short)0, cacc, false, false);
    }
    const float ksqn = KE * sqCol[ct * 16 + r];
#pragma unroll
    for (int v = 0; v < 8; ++v) {
      float arg = fminf(__builtin_fmaf(cacc[v], L2E, sb[v] + ksqn), 0.f);
      acc += EXP2F(arg);
    }
  }

  // --- block reduction -> one atomic per block (weight 2 for off-diagonal) ---
#pragma unroll
  for (int off = 16; off > 0; off >>= 1) acc += __shfl_xor(acc, off, 32);
  if (lane == 0) red[wave] = acc;
  __syncthreads();
  if (threadIdx.x == 0) {
    float tsum = 0.f;
#pragma unroll
    for (int w = 0; w < 8; ++w) tsum += red[w];
    atomicAdd(&yy_acc[b], (I == J) ? tsum : 2.f * tsum);
  }
}

// ---------------------------------------------------------------------------
// Kernel 3: finalize 32 outputs
// ---------------------------------------------------------------------------
__global__ void mmd_finalize_kernel(const float* __restrict__ yy_acc,
                                    const float* __restrict__ xy_acc,
                                    float* __restrict__ out) {
  const int b = threadIdx.x;
  if (b < NBATCH) {
    const float n        = (float)N_ROWS;
    const float x_kernel = 5.3966576851e-16f;   // (1/3)^32
    const float cfac     = 2.3283064365e-10f;   // (1/2)^32
    const float yy = yy_acc[b] / (n * (n - 1.f));
    const float xy = (2.f / n) * xy_acc[b] * cfac;
    out[b] = x_kernel + yy - xy;
  }
}

extern "C" void kernel_launch(void* const* d_in, const int* in_sizes, int n_in,
                              void* d_out, int out_size, void* d_ws, size_t ws_size,
                              hipStream_t stream) {
  (void)in_sizes; (void)n_in; (void)out_size; (void)ws_size;
  const float* x = (const float*)d_in[0];
  float* ws      = (float*)d_ws;
  float* sq      = ws;                        // 32*2048 floats
  float* yy_acc  = ws + NBATCH * N_ROWS;      // 32 floats
  float* xy_acc  = yy_acc + NBATCH;           // 32 floats

  (void)hipMemsetAsync(yy_acc, 0, 2 * NBATCH * sizeof(float), stream);
  mmd_sq_xy_kernel<<<dim3(8, NBATCH), 256, 0, stream>>>(x, sq, xy_acc);
  mmd_gram_exp_kernel<<<dim3(NPAIRS, NBATCH), 256, 0, stream>>>(x, sq, yy_acc);
  mmd_finalize_kernel<<<1, 32, 0, stream>>>(yy_acc, xy_acc, (float*)d_out);
}
